// TGCN_69380901700110
// MI455X (gfx1250) — matrix operand order repaired and checked by reading, not compiled, not run on hardware
//
#include <hip/hip_runtime.h>
#include <stdint.h>

// ---------------------------------------------------------------------------
// gfx1250 / CDNA5 primitives
// ---------------------------------------------------------------------------
#define USE_ASYNC_LDS 1

typedef __attribute__((ext_vector_type(16))) __bf16 v16bf;
typedef __attribute__((ext_vector_type(8)))  float  v8f;

#define FLAG_RELU     1
#define FLAG_BIAS_ROW 2
#define FLAG_BIAS_COL 4

__device__ __forceinline__ unsigned short f2bf(float f) {
  unsigned u = __builtin_bit_cast(unsigned, f);
  u += 0x7fffu + ((u >> 16) & 1u);           // round-to-nearest-even
  return (unsigned short)(u >> 16);
}
__device__ __forceinline__ float bf2f(unsigned short h) {
  unsigned u = ((unsigned)h) << 16;
  return __builtin_bit_cast(float, u);
}

union Frag {
  v16bf v;
  uint4 q[2];
};

// Stage 16B/lane from global into LDS via the CDNA5 async-to-LDS path (ASYNCcnt).
__device__ __forceinline__ void stage_b128(unsigned short* ldst, const void* gsrc) {
#if USE_ASYNC_LDS
  unsigned lds_addr = (unsigned)(uintptr_t)ldst;   // low 32 bits of generic LDS ptr = LDS byte address
  asm volatile("global_load_async_to_lds_b128 %0, %1, off"
               :: "v"(lds_addr), "v"((unsigned long long)(uintptr_t)gsrc)
               : "memory");
#else
  *(uint4*)ldst = *(const uint4*)gsrc;
#endif
}
__device__ __forceinline__ void wait_async() {
#if USE_ASYNC_LDS
  asm volatile("s_wait_asynccnt 0" ::: "memory");
#endif
}

// Force a wave-uniform condition into an SGPR so the compiler emits a scalar
// branch (EXEC stays all-1s inside -- required around WMMA).
__device__ __forceinline__ bool uniform(bool c) {
  return __builtin_amdgcn_readfirstlane(c ? 1 : 0) != 0;
}

// ---------------------------------------------------------------------------
// Generic WMMA GEMM:  D[M,N] = act( A[M,K] * Bt[N,K]^T + bias )
// A, Bt bf16 (K-major, 16B-aligned rows); K multiple of 32.
// Block = 8 waves; wave computes 16 rows x 64 cols (4 accumulators).
// Bt tile (64x32) double-buffered in LDS via async-to-LDS: the copy of the
// next K-chunk overlaps the 4 WMMAs of the current chunk; 1 barrier / chunk.
// ---------------------------------------------------------------------------
__global__ __launch_bounds__(256) void k_gemm_bt_wmma(
    const unsigned short* __restrict__ A, int lda,
    const unsigned short* __restrict__ Bt, int ldb,
    float* __restrict__ outF, unsigned short* __restrict__ outH, int ldc,
    int M, int N, int K,
    const float* __restrict__ bias, int flags)
{
  __shared__ unsigned short sB[2][64 * 32];
  const int t = threadIdx.x;
  const int lane = t & 31, wid = t >> 5;
  const int mtile = blockIdx.x * 128 + wid * 16;
  const int ntile = blockIdx.y * 64;
  const int row = lane & 15, khalf = lane >> 4;

  const int mA = min(mtile + row, M - 1);
  const unsigned short* Arow = A + (size_t)mA * lda;

  const int sn = t >> 2, sk = (t & 3) * 8;          // 256 threads cover 64x32
  const unsigned short* bsrc = Bt + (size_t)min(ntile + sn, N - 1) * ldb + sk;
  unsigned short* sdst[2] = { &sB[0][sn * 32 + sk], &sB[1][sn * 32 + sk] };

  v8f acc[4] = {};
  const bool wact = uniform(mtile < M);
  const int niter = K >> 5;

  stage_b128(sdst[0], bsrc);                         // prologue: chunk 0
  for (int i = 0; i < niter; ++i) {
    const int kc = i << 5;
    const int cur = i & 1;
    wait_async();                                    // our chunk-i piece landed
    __syncthreads();                                 // everyone's landed; prev reads done
    if (i + 1 < niter)
      stage_b128(sdst[cur ^ 1], bsrc + kc + 32);     // overlap copy of chunk i+1
    if (wact) {
      Frag a;   // A 16x32 bf16 fragment: lane row=lane&15, two 8-elt K-runs 16 apart
      a.q[0] = *(const uint4*)(Arow + kc + khalf * 8);
      a.q[1] = *(const uint4*)(Arow + kc + khalf * 8 + 16);
#pragma unroll
      for (int sub = 0; sub < 4; ++sub) {
        Frag b; // B 32x16: lane col=lane&15, two 8-elt K-runs 16 apart
        const unsigned short* bs = &sB[cur][(sub * 16 + row) * 32 + khalf * 8];
        b.q[0] = *(const uint4*)(bs);
        b.q[1] = *(const uint4*)(bs + 16);
        acc[sub] = __builtin_amdgcn_wmma_f32_16x16x32_bf16(
            false, a.v, false, b.v, (short)0, acc[sub], false, false);
      }
    }
  }
  if (!wact) return;
#pragma unroll
  for (int sub = 0; sub < 4; ++sub) {
    const int n = ntile + sub * 16 + row;
#pragma unroll
    for (int r = 0; r < 8; ++r) {
      const int m = mtile + r + 8 * khalf;
      if (m < M && n < N) {
        float v = acc[sub][r];
        if (flags & FLAG_BIAS_ROW) v += bias[m];
        if (flags & FLAG_BIAS_COL) v += bias[n];
        if (flags & FLAG_RELU) v = v > 0.f ? v : 0.f;
        const size_t o = (size_t)m * (size_t)ldc + n;
        if (outF) outF[o] = v;
        if (outH) outH[o] = f2bf(v);
      }
    }
  }
}

// ---------------------------------------------------------------------------
// Implicit-GEMM 3x3 SAME conv, HWC activations, bf16 in / fp32 acc.
// out[p, co] = sum_{kyx, ci} in[p+d(kyx), ci] * wrep[co, kyx, ci]
// M = pixels, N = Cout (mult of 64), K = 9*C (C mult of 32).
// Weight tiles double-buffered in LDS via async-to-LDS.
// Writes pre-activation (for side conv) and ReLU'd copy (for next conv).
// ---------------------------------------------------------------------------
__global__ __launch_bounds__(256) void k_conv3x3_wmma(
    const unsigned short* __restrict__ actin,
    const unsigned short* __restrict__ wrep,
    const float* __restrict__ bias,
    unsigned short* __restrict__ hpre,
    unsigned short* __restrict__ hrelu,
    int H, int W, int C, int Cout)
{
  __shared__ unsigned short sB[2][64 * 32];
  const int P = H * W;
  const int K = 9 * C;
  const int t = threadIdx.x;
  const int lane = t & 31, wid = t >> 5;
  const int ptile = blockIdx.x * 128 + wid * 16;
  const int cout0 = blockIdx.y * 64;
  const int row = lane & 15, khalf = lane >> 4;
  const int p = ptile + row;
  const int py = p / W, px = p - py * W;

  const int sn = t >> 2, sk = (t & 3) * 8;
  const unsigned short* wsrc = wrep + (size_t)(cout0 + sn) * K + sk;
  unsigned short* sdst[2] = { &sB[0][sn * 32 + sk], &sB[1][sn * 32 + sk] };

  v8f acc[4] = {};
  const bool wact = uniform(ptile < P);
  const int niter = K >> 5;

  stage_b128(sdst[0], wsrc);
  for (int i = 0; i < niter; ++i) {
    const int kc = i << 5;
    const int cur = i & 1;
    wait_async();
    __syncthreads();
    if (i + 1 < niter)
      stage_b128(sdst[cur ^ 1], wsrc + kc + 32);
    if (wact) {
      const int kyx = kc / C;                 // uniform across the 32-chunk (C % 32 == 0)
      const int ky = kyx / 3 - 1;
      const int kx = kyx - (kyx / 3) * 3 - 1;
      const int iy = py + ky, ix = px + kx;
      const bool valid = (p < P) & ((unsigned)iy < (unsigned)H) & ((unsigned)ix < (unsigned)W);
      Frag a;
      if (valid) {
        const unsigned short* ab =
            actin + (size_t)(iy * W + ix) * C + (kc - kyx * C) + khalf * 8;
        a.q[0] = *(const uint4*)(ab);
        a.q[1] = *(const uint4*)(ab + 16);
      } else {
        a.q[0] = make_uint4(0u, 0u, 0u, 0u);  // zero padding
        a.q[1] = make_uint4(0u, 0u, 0u, 0u);
      }
#pragma unroll
      for (int sub = 0; sub < 4; ++sub) {
        Frag b;
        const unsigned short* bs = &sB[cur][(sub * 16 + row) * 32 + khalf * 8];
        b.q[0] = *(const uint4*)(bs);
        b.q[1] = *(const uint4*)(bs + 16);
        acc[sub] = __builtin_amdgcn_wmma_f32_16x16x32_bf16(
            false, a.v, false, b.v, (short)0, acc[sub], false, false);
      }
    }
  }
  if (!wact) return;
#pragma unroll
  for (int sub = 0; sub < 4; ++sub) {
    const int cout = cout0 + sub * 16 + row;
    const float bv = bias[cout];
#pragma unroll
    for (int r = 0; r < 8; ++r) {
      const int pixel = ptile + r + 8 * khalf;
      if (pixel < P) {
        const float v = acc[sub][r] + bv;
        const size_t o = (size_t)pixel * Cout + cout;
        hpre[o]  = f2bf(v);
        hrelu[o] = f2bf(v > 0.f ? v : 0.f);
      }
    }
  }
}

// ---------------------------------------------------------------------------
// First conv (3->64) — tiny (173 MFLOP), plain kernel, fp32 in -> bf16 HWC out.
// ---------------------------------------------------------------------------
__global__ __launch_bounds__(256) void k_conv0(const float* __restrict__ x,
                                               const float* __restrict__ w,
                                               const float* __restrict__ bias,
                                               unsigned short* __restrict__ hpre,
                                               unsigned short* __restrict__ hrelu)
{
  const int H = 224, W = 224;
  long long idx = (long long)blockIdx.x * 256 + threadIdx.x;
  if (idx >= (long long)H * W * 64) return;
  int co = (int)(idx & 63);
  int p  = (int)(idx >> 6);
  int py = p / W, px = p - py * W;
  float s = bias[co];
  for (int ci = 0; ci < 3; ++ci)
#pragma unroll
    for (int ky = -1; ky <= 1; ++ky)
#pragma unroll
      for (int kx = -1; kx <= 1; ++kx) {
        int iy = py + ky, ix = px + kx;
        if ((unsigned)iy < (unsigned)H && (unsigned)ix < (unsigned)W)
          s += x[((size_t)ci * H + iy) * W + ix] *
               w[(((size_t)co * 3 + ci) * 3 + (ky + 1)) * 3 + (kx + 1)];
      }
  hpre[idx]  = f2bf(s);
  hrelu[idx] = f2bf(s > 0.f ? s : 0.f);
}

// ---------------------------------------------------------------------------
// 2x2 maxpool on HWC bf16.
// ---------------------------------------------------------------------------
__global__ __launch_bounds__(256) void k_maxpool(const unsigned short* __restrict__ in,
                                                 unsigned short* __restrict__ out,
                                                 int Ho, int Wo, int C)
{
  long long idx = (long long)blockIdx.x * 256 + threadIdx.x;
  long long total = (long long)Ho * Wo * C;
  if (idx >= total) return;
  int c  = (int)(idx % C);
  long long q = idx / C;
  int xo = (int)(q % Wo);
  int yo = (int)(q / Wo);
  int Wi = Wo * 2;
  const unsigned short* b = in + ((size_t)(yo * 2) * Wi + xo * 2) * C + c;
  float m = fmaxf(fmaxf(bf2f(b[0]), bf2f(b[C])),
                  fmaxf(bf2f(b[(size_t)Wi * C]), bf2f(b[(size_t)Wi * C + C])));
  out[idx] = f2bf(m);
}

// ---------------------------------------------------------------------------
// Weight repack: OIHW fp32 -> [co][kyx][ci] bf16 (K-major for the conv GEMM).
// ---------------------------------------------------------------------------
__global__ __launch_bounds__(256) void k_repack_w3(const float* __restrict__ src,
                                                   unsigned short* __restrict__ dst,
                                                   int C, int Cout)
{
  long long idx = (long long)blockIdx.x * 256 + threadIdx.x;
  long long total = (long long)Cout * 9 * C;
  if (idx >= total) return;
  int co  = (int)(idx / (9 * (long long)C));
  int rem = (int)(idx - (long long)co * 9 * C);
  int kyx = rem / C;
  int ci  = rem - kyx * C;
  dst[idx] = f2bf(src[((size_t)co * C + ci) * 9 + kyx]);
}

__global__ __launch_bounds__(256) void k_cast_bf16(const float* __restrict__ src,
                                                   unsigned short* __restrict__ dst,
                                                   long long n)
{
  long long i = (long long)blockIdx.x * 256 + threadIdx.x;
  if (i < n) dst[i] = f2bf(src[i]);
}

// ---------------------------------------------------------------------------
// Superpixel-map downsampling through the transposed bilinear-resize operator.
// mat[i][O] = resize weight of output sample O (of 224) onto source index i (of h).
// ---------------------------------------------------------------------------
__global__ __launch_bounds__(256) void k_build_rmat(float* __restrict__ mat, int h)
{
  long long idx = (long long)blockIdx.x * 256 + threadIdx.x;
  if (idx >= (long long)h * 224) return;
  int i = (int)(idx / 224);
  int O = (int)(idx - (long long)i * 224);
  float s = (float)O * (float)(h - 1) / 223.0f;   // linspace(0, h-1, 224)
  int i0 = (int)floorf(s);
  int i1 = min(i0 + 1, h - 1);
  float w = s - (float)i0;
  float v = 0.f;
  if (i == i0) v += 1.f - w;
  if (i == i1) v += w;
  mat[idx] = v;
}

// tmp[s*224+Y][x] = sum_X sp[s,Y,X] * mat[x][X]
__global__ __launch_bounds__(256) void k_sp_tmp(const float* __restrict__ sp,
                                                const float* __restrict__ mat,
                                                float* __restrict__ tmp, int w)
{
  long long idx = (long long)blockIdx.x * 256 + threadIdx.x;
  long long total = 256LL * 224 * w;
  if (idx >= total) return;
  int x = (int)(idx % w);
  long long sY = idx / w;
  const float* sprow = sp + (size_t)sY * 224;
  const float* mrow  = mat + (size_t)x * 224;
  float s = 0.f;
  for (int X = 0; X < 224; ++X) s += sprow[X] * mrow[X];
  tmp[idx] = s;
}

// spd[s][k] (bf16, K padded to mult-of-32, zero tail) = sum_Y mat[y][Y]*tmp[s*224+Y][x]
__global__ __launch_bounds__(256) void k_sp_down(const float* __restrict__ tmp,
                                                 const float* __restrict__ mat,
                                                 unsigned short* __restrict__ spd,
                                                 int h, int w, int Kpad)
{
  long long idx = (long long)blockIdx.x * 256 + threadIdx.x;
  long long total = 256LL * Kpad;
  if (idx >= total) return;
  int k = (int)(idx % Kpad);
  int s = (int)(idx / Kpad);
  float v = 0.f;
  if (k < h * w) {
    int y = k / w, x = k - y * w;
    const float* trow = tmp + (size_t)s * 224 * w + x;
    const float* mrow = mat + (size_t)y * 224;
    for (int Y = 0; Y < 224; ++Y) v += mrow[Y] * trow[(size_t)Y * w];
  }
  spd[idx] = f2bf(v);
}

// ---------------------------------------------------------------------------
// softmax over the 2 classes; keep only P(class 1).
// ---------------------------------------------------------------------------
__global__ void k_softmax2(const float* __restrict__ logits, float* __restrict__ prob1)
{
  int s = blockIdx.x * blockDim.x + threadIdx.x;
  if (s >= 256) return;
  float a = logits[s * 2 + 0], b = logits[s * 2 + 1];
  float m = fmaxf(a, b);
  float ea = __expf(a - m), eb = __expf(b - m);
  prob1[s] = eb / (ea + eb);
}

// per-pixel argmax over the 256 superpixel maps, gather P1.
__global__ __launch_bounds__(256) void k_assign(const float* __restrict__ sp,
                                                const float* __restrict__ prob1,
                                                float* __restrict__ out)
{
  int p = blockIdx.x * 256 + threadIdx.x;
  if (p >= 224 * 224) return;
  float best = sp[p];
  int bi = 0;
  for (int s = 1; s < 256; ++s) {
    if (s + 8 < 256) __builtin_prefetch(sp + (size_t)(s + 8) * 50176 + p, 0, 0);
    float v = sp[(size_t)s * 50176 + p];
    if (v > best) { best = v; bi = s; }
  }
  out[p] = prob1[bi];
}

// ---------------------------------------------------------------------------
// Host orchestration
// ---------------------------------------------------------------------------
static void gemm(hipStream_t stream, const unsigned short* A, int lda,
                 const unsigned short* Bt, int ldb, float* outF, unsigned short* outH,
                 int ldc, int M, int N, int K, const float* bias, int flags)
{
  dim3 grid((unsigned)((M + 127) / 128), (unsigned)((N + 63) / 64));
  k_gemm_bt_wmma<<<grid, dim3(256), 0, stream>>>(A, lda, Bt, ldb, outF, outH, ldc,
                                                 M, N, K, bias, flags);
}

static inline unsigned elgrid(long long n) { return (unsigned)((n + 255) / 256); }

extern "C" void kernel_launch(void* const* d_in, const int* in_sizes, int n_in,
                              void* d_out, int out_size, void* d_ws, size_t ws_size,
                              hipStream_t stream)
{
  (void)in_sizes; (void)n_in; (void)out_size; (void)ws_size;

  static const int Lcin[13]  = {3, 64, 64, 128, 128, 256, 256, 256, 512, 512, 512, 512, 512};
  static const int Lcout[13] = {64, 64, 128, 128, 256, 256, 256, 512, 512, 512, 512, 512, 512};
  static const int Lres[13]  = {224, 224, 112, 112, 56, 56, 56, 28, 28, 28, 14, 14, 14};
  static const bool Lpool[13] = {false,true,false,true,false,false,true,false,false,true,false,false,false};

  const float* x  = (const float*)d_in[0];
  const float* sp = (const float*)d_in[1];
  const float *convW[13], *convB[13], *sideW[13], *sideB[13];
  for (int l = 0; l < 13; ++l) {
    convW[l] = (const float*)d_in[2  + 2 * l];
    convB[l] = (const float*)d_in[3  + 2 * l];
    sideW[l] = (const float*)d_in[28 + 2 * l];
    sideB[l] = (const float*)d_in[29 + 2 * l];
  }
  const float *fcW[7], *fcB[7];
  for (int j = 0; j < 3; ++j) { fcW[j]     = (const float*)d_in[54 + 2 * j]; fcB[j]     = (const float*)d_in[55 + 2 * j]; }
  for (int j = 0; j < 3; ++j) { fcW[3 + j] = (const float*)d_in[60 + 2 * j]; fcB[3 + j] = (const float*)d_in[61 + 2 * j]; }
  fcW[6] = (const float*)d_in[66]; fcB[6] = (const float*)d_in[67];
  static const int fcN[7] = {512, 512, 12, 512, 512, 20, 2};
  static const int fcK[7] = {1056, 512, 512, 1056, 512, 512, 32};

  // ---- workspace bump allocator --------------------------------------------
  char* wsb = (char*)d_ws;
  size_t wsoff = 0;
  auto alloc = [&](size_t bytes) -> void* {
    void* p = wsb + wsoff;
    wsoff = (wsoff + bytes + 255) & ~(size_t)255;
    return p;
  };

  unsigned short* wrep[13]; wrep[0] = nullptr;
  for (int l = 1; l < 13; ++l)
    wrep[l] = (unsigned short*)alloc((size_t)9 * Lcin[l] * Lcout[l] * 2);
  unsigned short* wsd[13];
  for (int l = 0; l < 13; ++l)
    wsd[l] = (unsigned short*)alloc((size_t)(Lcout[l] / 2) * Lcout[l] * 2);
  unsigned short* fcWH[7];
  for (int j = 0; j < 7; ++j)
    fcWH[j] = (unsigned short*)alloc((size_t)fcN[j] * fcK[j] * 2);

  unsigned short* actPre = (unsigned short*)alloc((size_t)50176 * 64 * 2);
  unsigned short* actA   = (unsigned short*)alloc((size_t)50176 * 64 * 2);
  unsigned short* actB   = (unsigned short*)alloc((size_t)50176 * 64 * 2);

  unsigned short* fmb[13];
  int Kpad[13], C2[13], coloff[13];
  int co = 0;
  for (int l = 0; l < 13; ++l) {
    int P = Lres[l] * Lres[l];
    Kpad[l] = ((P + 31) / 32) * 32;
    C2[l] = Lcout[l] / 2;
    coloff[l] = co; co += C2[l];
    fmb[l] = (unsigned short*)alloc((size_t)C2[l] * Kpad[l] * 2);
  }

  unsigned short* sp224 = (unsigned short*)alloc((size_t)256 * 50176 * 2);
  float* rmat  = (float*)alloc((size_t)112 * 224 * 4);
  float* tmpb  = (float*)alloc((size_t)256 * 224 * 112 * 4);
  unsigned short* spd = (unsigned short*)alloc((size_t)256 * 12544 * 2);
  float*          pooledF = (float*)alloc((size_t)256 * 2112 * 4);
  unsigned short* pooledH = (unsigned short*)alloc((size_t)256 * 2112 * 2);
  unsigned short* mh1    = (unsigned short*)alloc((size_t)256 * 512 * 2);
  unsigned short* mh2    = (unsigned short*)alloc((size_t)256 * 512 * 2);
  unsigned short* spfeat = (unsigned short*)alloc((size_t)256 * 32 * 2);
  float* logits = (float*)alloc((size_t)256 * 2 * 4);
  float* prob1  = (float*)alloc((size_t)256 * 4);

  // ---- weight conversion / repack ------------------------------------------
  for (int l = 1; l < 13; ++l) {
    long long n = (long long)Lcout[l] * Lcin[l] * 9;
    k_repack_w3<<<dim3(elgrid(n)), dim3(256), 0, stream>>>(convW[l], wrep[l], Lcin[l], Lcout[l]);
  }
  for (int l = 0; l < 13; ++l) {
    long long n = (long long)C2[l] * Lcout[l];
    k_cast_bf16<<<dim3(elgrid(n)), dim3(256), 0, stream>>>(sideW[l], wsd[l], n);
  }
  for (int j = 0; j < 7; ++j) {
    long long n = (long long)fcN[j] * fcK[j];
    k_cast_bf16<<<dim3(elgrid(n)), dim3(256), 0, stream>>>(fcW[j], fcWH[j], n);
  }

  // ---- VGG trunk + side heads ----------------------------------------------
  k_conv0<<<dim3(elgrid(50176LL * 64)), dim3(256), 0, stream>>>(x, convW[0], convB[0], actPre, actA);
  gemm(stream, wsd[0], 64, actPre, 64, nullptr, fmb[0], Kpad[0],
       C2[0], 50176, 64, sideB[0], FLAG_BIAS_ROW);

  unsigned short* cur = actA;
  unsigned short* oth = actB;
  for (int l = 1; l < 13; ++l) {
    int R = Lres[l], P = R * R, cin = Lcin[l], cout = Lcout[l];
    dim3 g((unsigned)((P + 127) / 128), (unsigned)(cout / 64));
    k_conv3x3_wmma<<<g, dim3(256), 0, stream>>>(cur, wrep[l], convB[l], actPre, oth,
                                                R, R, cin, cout);
    gemm(stream, wsd[l], cout, actPre, cout, nullptr, fmb[l], Kpad[l],
         C2[l], P, cout, sideB[l], FLAG_BIAS_ROW);
    { unsigned short* t2 = cur; cur = oth; oth = t2; }
    if (Lpool[l]) {
      int Ho = R / 2;
      long long n = (long long)Ho * Ho * cout;
      k_maxpool<<<dim3(elgrid(n)), dim3(256), 0, stream>>>(cur, oth, Ho, Ho, cout);
      unsigned short* t2 = cur; cur = oth; oth = t2;
    }
  }

  // ---- superpixel pooling: pooled[s,c] = <R^T sp[s], fm_c> at native res ----
  k_cast_bf16<<<dim3(elgrid(256LL * 50176)), dim3(256), 0, stream>>>(sp, sp224, 256LL * 50176);
  for (int l = 0; l < 2; ++l)   // 224x224 layers: identity resize
    gemm(stream, sp224, 50176, fmb[l], Kpad[l], pooledF + coloff[l], pooledH + coloff[l],
         2112, 256, C2[l], 50176, nullptr, 0);

  const int gh[4]  = {112, 56, 28, 14};
  const int gl0[4] = {2, 4, 7, 10};
  const int gl1[4] = {3, 6, 9, 12};
  for (int gi = 0; gi < 4; ++gi) {
    int h = gh[gi];
    int Kp = Kpad[gl0[gi]];
    k_build_rmat<<<dim3(elgrid((long long)h * 224)), dim3(256), 0, stream>>>(rmat, h);
    k_sp_tmp<<<dim3(elgrid(256LL * 224 * h)), dim3(256), 0, stream>>>(sp, rmat, tmpb, h);
    k_sp_down<<<dim3(elgrid(256LL * Kp)), dim3(256), 0, stream>>>(tmpb, rmat, spd, h, h, Kp);
    for (int l = gl0[gi]; l <= gl1[gi]; ++l)
      gemm(stream, spd, Kp, fmb[l], Kp, pooledF + coloff[l], pooledH + coloff[l],
           2112, 256, C2[l], Kp, nullptr, 0);
  }

  // ---- MLPs + classifier ----------------------------------------------------
  gemm(stream, pooledH,        2112, fcWH[0], 1056, nullptr, mh1, 512, 256, 512, 1056, fcB[0], FLAG_BIAS_COL | FLAG_RELU);
  gemm(stream, mh1,             512, fcWH[1],  512, nullptr, mh2, 512, 256, 512,  512, fcB[1], FLAG_BIAS_COL | FLAG_RELU);
  gemm(stream, mh2,             512, fcWH[2],  512, nullptr, spfeat,      32, 256, 12, 512, fcB[2], FLAG_BIAS_COL | FLAG_RELU);
  gemm(stream, pooledH + 1056, 2112, fcWH[3], 1056, nullptr, mh1, 512, 256, 512, 1056, fcB[3], FLAG_BIAS_COL | FLAG_RELU);
  gemm(stream, mh1,             512, fcWH[4],  512, nullptr, mh2, 512, 256, 512,  512, fcB[4], FLAG_BIAS_COL | FLAG_RELU);
  gemm(stream, mh2,             512, fcWH[5],  512, nullptr, spfeat + 12, 32, 256, 20, 512, fcB[5], FLAG_BIAS_COL | FLAG_RELU);
  gemm(stream, spfeat,           32, fcWH[6],   32, logits, nullptr, 2, 256, 2, 32, fcB[6], FLAG_BIAS_COL);

  k_softmax2<<<dim3(1), dim3(256), 0, stream>>>(logits, prob1);
  k_assign<<<dim3(elgrid(50176)), dim3(256), 0, stream>>>(sp, prob1, (float*)d_out);
}
